// LinearGlobalAttn_87230785781976
// MI455X (gfx1250) — matrix operand order repaired and checked by reading, not compiled, not run on hardware
//
#include <hip/hip_runtime.h>
#include <hip/hip_bf16.h>
#include <math.h>

typedef _Float16 v16h __attribute__((ext_vector_type(16)));
typedef _Float16 v8h  __attribute__((ext_vector_type(8)));
typedef float    v8f  __attribute__((ext_vector_type(8)));

#define LN_EPS 1e-5f
#define CLAMPV 100.0f

// Problem dims (fixed by the reference)
static constexpr int  Bx = 8;          // batch
static constexpr int  Nt = 4096;       // tokens (= 64*64)
static constexpr int  Cc = 512;        // channels
static constexpr long long QKVLD = 1536; // qkv row stride

// ---------------------------------------------------------------------------
// Generic batched WMMA GEMM: C = op(A) * B (+bias), f16 inputs, f32 accum.
// A: [M x K] (or K x M if transA) row-major with leading dim lda
// B: [K x N] row-major, ldb.   C: [M x N], ldc, f32 or f16 output.
// Batch z decomposed as (z/HB, z%HB) with two strides per matrix.
// All operand base addresses / strides are 16-byte aligned for the shapes we
// launch, so tile loads go through 128-bit vector paths with a guarded scalar
// fallback for partial tiles.
// ---------------------------------------------------------------------------
template<int BM, int BN, int BK, int WR, int WC>
__global__ __launch_bounds__(WR*WC*32)
void gemm_wmma_kernel(const _Float16* __restrict__ A,
                      const _Float16* __restrict__ Bmat,
                      const float* __restrict__ bias,
                      void* __restrict__ Cout, int outHalf,
                      int M, int N, int K,
                      int lda, int ldb, int ldc,
                      int transA, int reluA, int reluB,
                      long long sA1, long long sA2,
                      long long sB1, long long sB2,
                      long long sC1, long long sC2, int HB)
{
    constexpr int THREADS = WR*WC*32;
    constexpr int MT = BM/(WR*16);
    constexpr int NT = BN/(WC*16);
    constexpr int AV = BM*BK/8;       // 128-bit vectors per A tile
    constexpr int BV = BN*BK/8;       // 128-bit vectors per B tile
    __shared__ _Float16 As[BM][BK];
    __shared__ _Float16 Bs[BN][BK];   // stored column(n)-major: Bs[n][k]

    const int tid    = threadIdx.x;
    const int wid    = tid >> 5;
    const int lane   = tid & 31;
    const int half16 = lane >> 4;
    const int r      = lane & 15;
    const int wm     = wid / WC;
    const int wn     = wid % WC;
    const int m0     = blockIdx.y * BM;
    const int n0     = blockIdx.x * BN;
    const int z      = blockIdx.z;
    const long long aoff = (long long)(z / HB) * sA1 + (long long)(z % HB) * sA2;
    const long long boff = (long long)(z / HB) * sB1 + (long long)(z % HB) * sB2;
    const long long coff = (long long)(z / HB) * sC1 + (long long)(z % HB) * sC2;

    v8f acc[MT][NT];
    #pragma unroll
    for (int mt = 0; mt < MT; ++mt)
        #pragma unroll
        for (int nt = 0; nt < NT; ++nt)
            #pragma unroll
            for (int e = 0; e < 8; ++e) acc[mt][nt][e] = 0.0f;

    for (int k0 = 0; k0 < K; k0 += BK) {
        // ---- load A tile into LDS (128-bit global loads) ----
        if (!transA) {
            for (int idx = tid; idx < AV; idx += THREADS) {
                int row = idx / (BK/8);
                int kk8 = (idx % (BK/8)) * 8;
                int gm = m0 + row, gk = k0 + kk8;
                v8h v;
                if (gm < M && gk + 8 <= K) {
                    v = *(const v8h*)(A + aoff + (long long)gm*lda + gk);
                } else {
                    #pragma unroll
                    for (int e = 0; e < 8; ++e)
                        v[e] = (gm < M && gk + e < K)
                             ? A[aoff + (long long)gm*lda + gk + e] : (_Float16)0.0f;
                }
                if (reluA) {
                    #pragma unroll
                    for (int e = 0; e < 8; ++e)
                        v[e] = v[e] > (_Float16)0.0f ? v[e] : (_Float16)0.0f;
                }
                *(v8h*)&As[row][kk8] = v;   // ds_store_b128
            }
        } else {
            // A is K x M: contiguous along m -> vector load, scatter to LDS
            for (int idx = tid; idx < AV; idx += THREADS) {
                int kk = idx / (BM/8);
                int m8 = (idx % (BM/8)) * 8;
                int gk = k0 + kk, gm = m0 + m8;
                v8h v;
                if (gk < K && gm + 8 <= M) {
                    v = *(const v8h*)(A + aoff + (long long)gk*lda + gm);
                } else {
                    #pragma unroll
                    for (int e = 0; e < 8; ++e)
                        v[e] = (gk < K && gm + e < M)
                             ? A[aoff + (long long)gk*lda + gm + e] : (_Float16)0.0f;
                }
                if (reluA) {
                    #pragma unroll
                    for (int e = 0; e < 8; ++e)
                        v[e] = v[e] > (_Float16)0.0f ? v[e] : (_Float16)0.0f;
                }
                #pragma unroll
                for (int e = 0; e < 8; ++e) As[m8 + e][kk] = v[e];
            }
        }
        // ---- load B tile (K x N, contiguous along n) -> Bs[n][k] ----
        for (int idx = tid; idx < BV; idx += THREADS) {
            int kk = idx / (BN/8);
            int n8 = (idx % (BN/8)) * 8;
            int gk = k0 + kk, gn = n0 + n8;
            v8h v;
            if (gk < K && gn + 8 <= N) {
                v = *(const v8h*)(Bmat + boff + (long long)gk*ldb + gn);
            } else {
                #pragma unroll
                for (int e = 0; e < 8; ++e)
                    v[e] = (gk < K && gn + e < N)
                         ? Bmat[boff + (long long)gk*ldb + gn + e] : (_Float16)0.0f;
            }
            if (reluB) {
                #pragma unroll
                for (int e = 0; e < 8; ++e)
                    v[e] = v[e] > (_Float16)0.0f ? v[e] : (_Float16)0.0f;
            }
            #pragma unroll
            for (int e = 0; e < 8; ++e) Bs[n8 + e][kk] = v[e];
        }
        // prefetch next A tile (global_prefetch_b8 on gfx1250)
        if (k0 + BK < K) {
            long long pr = transA
                ? (aoff + (long long)(k0 + BK)*lda + (m0 + (tid % BM)))
                : (aoff + (long long)(m0 + (tid % BM))*lda + (k0 + BK));
            __builtin_prefetch((const void*)(A + pr), 0, 3);
        }
        __syncthreads();

        #pragma unroll
        for (int kk = 0; kk < BK; kk += 32) {
            union { v16h v; v8h h[2]; } af[MT], bf[NT];
            // A 16x32 f16 layout: half 0 -> K{0..7,16..23}, half 1 -> K{8..15,24..31}
            #pragma unroll
            for (int mt = 0; mt < MT; ++mt) {
                int row = wm*(BM/WR) + mt*16 + r;
                af[mt].h[0] = *(const v8h*)&As[row][kk + half16*8];
                af[mt].h[1] = *(const v8h*)&As[row][kk + 16 + half16*8];
            }
            // B 32x16 f16 layout: half 0 -> K{0..7,16..23}, half 1 -> K{8..15,24..31}
            #pragma unroll
            for (int nt = 0; nt < NT; ++nt) {
                int col = wn*(BN/WC) + nt*16 + r;
                bf[nt].h[0] = *(const v8h*)&Bs[col][kk + half16*8];
                bf[nt].h[1] = *(const v8h*)&Bs[col][kk + 16 + half16*8];
            }
            #pragma unroll
            for (int mt = 0; mt < MT; ++mt)
                #pragma unroll
                for (int nt = 0; nt < NT; ++nt)
                    acc[mt][nt] = __builtin_amdgcn_wmma_f32_16x16x32_f16(
                        false, af[mt].v, false, bf[nt].v,
                        (short)0, acc[mt][nt], false, false);
        }
        __syncthreads();
    }

    // ---- epilogue: C/D layout -> VGPR i: M = 8*half + i, N = lane%16 ----
    #pragma unroll
    for (int mt = 0; mt < MT; ++mt) {
        #pragma unroll
        for (int nt = 0; nt < NT; ++nt) {
            int gc = n0 + wn*(BN/WC) + nt*16 + r;
            if (gc >= N) continue;
            float bv = bias ? bias[gc] : 0.0f;
            #pragma unroll
            for (int i = 0; i < 8; ++i) {
                int gm = m0 + wm*(BM/WR) + mt*16 + half16*8 + i;
                if (gm >= M) continue;
                float v = acc[mt][nt][i] + bv;
                long long ci = coff + (long long)gm*ldc + gc;
                if (outHalf) ((_Float16*)Cout)[ci] = (_Float16)v;
                else         ((float*)Cout)[ci]    = v;
            }
        }
    }
}

// ---------------------------------------------------------------------------
// f32 -> f16 conversion (weights, kv)
// ---------------------------------------------------------------------------
__global__ void cvt_f32_f16_kernel(const float* __restrict__ s,
                                   _Float16* __restrict__ d, int n)
{
    int i = blockIdx.x * blockDim.x + threadIdx.x;
    if (i < n) d[i] = (_Float16)s[i];
}

// ---------------------------------------------------------------------------
// LayerNorm over C=512; outputs either f16 row-major or f32 transposed [B,C,N]
// ---------------------------------------------------------------------------
__global__ __launch_bounds__(256)
void layernorm_kernel(const float* __restrict__ in,
                      const float* __restrict__ g, const float* __restrict__ b,
                      _Float16* __restrict__ out_h, float* __restrict__ out_f,
                      int transposeOut)
{
    __shared__ float r1[256], r2[256];
    int t = threadIdx.x;
    long long base = (long long)blockIdx.x * Cc;
    float v0 = in[base + t], v1 = in[base + t + 256];
    r1[t] = v0 + v1; r2[t] = v0*v0 + v1*v1;
    __syncthreads();
    for (int s = 128; s > 0; s >>= 1) {
        if (t < s) { r1[t] += r1[t+s]; r2[t] += r2[t+s]; }
        __syncthreads();
    }
    float mean = r1[0] * (1.0f/Cc);
    float var  = r2[0] * (1.0f/Cc) - mean*mean;
    float rstd = rsqrtf(var + LN_EPS);
    float vals[2] = {v0, v1};
    #pragma unroll
    for (int ii = 0; ii < 2; ++ii) {
        int c = t + ii*256;
        float v = (vals[ii] - mean) * rstd * g[c] + b[c];
        if (transposeOut) {
            int bb = blockIdx.x / Nt, p = blockIdx.x % Nt;
            out_f[((long long)bb*Cc + c)*Nt + p] = v;
        } else {
            out_h[base + c] = (_Float16)v;
        }
    }
}

// ---------------------------------------------------------------------------
// normalizer[b,h,d] = max( sum_n relu(k[b,n,h*64+d]), 100 )
// ---------------------------------------------------------------------------
__global__ __launch_bounds__(256)
void ksum_kernel(const _Float16* __restrict__ qkv, float* __restrict__ normzr)
{
    __shared__ float sd[256];
    int t = threadIdx.x;
    int d = t & 63, seg = t >> 6;
    int b = blockIdx.x >> 3, h = blockIdx.x & 7;
    long long base = (long long)b*Nt*QKVLD + 512 + h*64 + d;
    float acc = 0.0f;
    for (int n = seg; n < Nt; n += 4)
        acc += fmaxf((float)qkv[base + (long long)n*QKVLD], 0.0f);
    sd[t] = acc;
    __syncthreads();
    if (seg == 0) {
        float tot = sd[d] + sd[d+64] + sd[d+128] + sd[d+192];
        normzr[blockIdx.x*64 + d] = fmaxf(tot, CLAMPV);
    }
}

// ---------------------------------------------------------------------------
// attn_scaled[b,n,c] = attn1[b,h,n,d] / max(sum_d relu(q)*normzr, 100)  (f16)
// ---------------------------------------------------------------------------
__global__ __launch_bounds__(512)
void attn_scale_kernel(const _Float16* __restrict__ qkv,
                       const float* __restrict__ attn1,
                       const float* __restrict__ normzr,
                       _Float16* __restrict__ out)
{
    __shared__ float sd[512];
    int t = threadIdx.x;
    int row = blockIdx.x;            // b*4096 + n
    int b = row >> 12, n = row & 4095;
    int h = t >> 6, d = t & 63;
    float q = fmaxf((float)qkv[(long long)row*QKVLD + t], 0.0f);
    sd[t] = q * normzr[(b*8 + h)*64 + d];
    __syncthreads();
    for (int s = 32; s > 0; s >>= 1) {
        if (d < s) sd[t] += sd[t + s];
        __syncthreads();
    }
    float denom = fmaxf(sd[h*64], CLAMPV);
    float v = attn1[((long long)(b*8 + h)*Nt + n)*64 + d] / denom;
    out[(long long)row*Cc + t] = (_Float16)v;
}

// ---------------------------------------------------------------------------
// depthwise 7x7 conv + bias + exact GELU; deterministic per-block partial stats
// grid: (16 tiles, C, B); block: 256 = 16x16 outputs; LDS halo 22x22
// ---------------------------------------------------------------------------
__global__ __launch_bounds__(256)
void dwconv_gelu_kernel(const float* __restrict__ in, const float* __restrict__ w,
                        const float* __restrict__ bias, float* __restrict__ out,
                        float* __restrict__ partials)
{
    __shared__ float tile[22*22];
    __shared__ float wt[49];
    __shared__ float red[256];
    int tid = threadIdx.x;
    int tileIdx = blockIdx.x, c = blockIdx.y, bz = blockIdx.z;
    int ty0 = (tileIdx >> 2) * 16, tx0 = (tileIdx & 3) * 16;
    const float* ip = in + ((long long)bz*Cc + c) * Nt;
    for (int idx = tid; idx < 484; idx += 256) {
        int iy = idx / 22 + ty0 - 3, ix = idx % 22 + tx0 - 3;
        tile[idx] = (iy >= 0 && iy < 64 && ix >= 0 && ix < 64) ? ip[iy*64 + ix] : 0.0f;
    }
    if (tid < 49) wt[tid] = w[c*49 + tid];
    __syncthreads();
    int ty = tid >> 4, tx = tid & 15;
    float acc = bias[c];
    #pragma unroll
    for (int i = 0; i < 7; ++i)
        #pragma unroll
        for (int j = 0; j < 7; ++j)
            acc = fmaf(wt[i*7 + j], tile[(ty + i)*22 + tx + j], acc);
    float g = 0.5f * acc * (1.0f + erff(acc * 0.70710678118654752f));
    out[((long long)bz*Cc + c)*Nt + (ty0 + ty)*64 + tx0 + tx] = g;

    red[tid] = g;       __syncthreads();
    for (int s = 128; s > 0; s >>= 1) { if (tid < s) red[tid] += red[tid+s]; __syncthreads(); }
    float ssum = red[0]; __syncthreads();
    red[tid] = g*g;     __syncthreads();
    for (int s = 128; s > 0; s >>= 1) { if (tid < s) red[tid] += red[tid+s]; __syncthreads(); }
    if (tid == 0) {
        int pb = bz*16 + tileIdx;   // 128 partials per channel
        partials[((long long)c*128 + pb)*2 + 0] = ssum;
        partials[((long long)c*128 + pb)*2 + 1] = red[0];
    }
}

__global__ void bn_finalize_kernel(const float* __restrict__ partials,
                                   const float* __restrict__ bn_g,
                                   const float* __restrict__ bn_b,
                                   float* __restrict__ bnss)
{
    int c = threadIdx.x;
    if (c >= Cc) return;
    float s = 0.0f, s2 = 0.0f;
    for (int i = 0; i < 128; ++i) {
        s  += partials[((long long)c*128 + i)*2 + 0];
        s2 += partials[((long long)c*128 + i)*2 + 1];
    }
    const float inv = 1.0f / (float)(Bx * Nt);
    float m = s * inv;
    float v = s2 * inv - m*m;
    float sc = bn_g[c] * rsqrtf(v + LN_EPS);
    bnss[c]       = sc;
    bnss[c + Cc]  = bn_b[c] - m * sc;
}

__global__ void bn_apply_kernel(const float* __restrict__ y,
                                const float* __restrict__ bnss,
                                float* __restrict__ y2)
{
    long long idx = (long long)blockIdx.x * blockDim.x + threadIdx.x;
    int c = (int)((idx >> 12) & 511);
    y2[idx] = y[idx] * bnss[c] + bnss[c + Cc];
}

// ---------------------------------------------------------------------------
// second depthwise conv + final residual: out[b,n,c] = x + attn_proj + conv
// ---------------------------------------------------------------------------
__global__ __launch_bounds__(256)
void dwconv_residual_kernel(const float* __restrict__ in, const float* __restrict__ w,
                            const float* __restrict__ bias,
                            const float* __restrict__ x, const float* __restrict__ proj,
                            float* __restrict__ out)
{
    __shared__ float tile[22*22];
    __shared__ float wt[49];
    int tid = threadIdx.x;
    int tileIdx = blockIdx.x, c = blockIdx.y, bz = blockIdx.z;
    int ty0 = (tileIdx >> 2) * 16, tx0 = (tileIdx & 3) * 16;
    const float* ip = in + ((long long)bz*Cc + c) * Nt;
    for (int idx = tid; idx < 484; idx += 256) {
        int iy = idx / 22 + ty0 - 3, ix = idx % 22 + tx0 - 3;
        tile[idx] = (iy >= 0 && iy < 64 && ix >= 0 && ix < 64) ? ip[iy*64 + ix] : 0.0f;
    }
    if (tid < 49) wt[tid] = w[c*49 + tid];
    __syncthreads();
    int ty = tid >> 4, tx = tid & 15;
    float acc = bias[c];
    #pragma unroll
    for (int i = 0; i < 7; ++i)
        #pragma unroll
        for (int j = 0; j < 7; ++j)
            acc = fmaf(wt[i*7 + j], tile[(ty + i)*22 + tx + j], acc);
    int p = (ty0 + ty)*64 + tx0 + tx;
    long long xi = ((long long)bz*Nt + p)*Cc + c;
    out[xi] = x[xi] + proj[xi] + acc;
}

// ---------------------------------------------------------------------------
// Workspace layout (bytes).  Regions are reused once their producer is dead.
// ---------------------------------------------------------------------------
static constexpr size_t OFF_NORMED  = 0;                         // 32MB f16; reused: attn_scaled
static constexpr size_t OFF_SMALL   = 32ull  << 20;
static constexpr size_t OFF_QKVW_H  = OFF_SMALL;                 // 1.5MB
static constexpr size_t OFF_OUTW_H  = OFF_SMALL + 1572864;       // 0.5MB
static constexpr size_t OFF_KV_F    = OFF_SMALL + 2097152;       // 1MB
static constexpr size_t OFF_KV_H    = OFF_SMALL + 3145728;       // 0.5MB
static constexpr size_t OFF_NORMZR  = OFF_SMALL + 3670016;       // 16KB
static constexpr size_t OFF_PART    = OFF_SMALL + 3686400;       // 512KB
static constexpr size_t OFF_BNSS    = OFF_SMALL + 4210688;       // 4KB
static constexpr size_t OFF_QKV_H   = 40ull  << 20;              // 96MB f16; reused: xi, then y2
static constexpr size_t OFF_XI      = OFF_QKV_H;                 // 64MB f32
static constexpr size_t OFF_Y2      = OFF_QKV_H;                 // 64MB f32 (after xi dead)
static constexpr size_t OFF_ATTN1   = 136ull << 20;              // 64MB f32; reused: y
static constexpr size_t OFF_Y       = OFF_ATTN1;                 // 64MB f32 (after attn1 dead)
static constexpr size_t OFF_PROJ    = 200ull << 20;              // 64MB f32 (live to end)

extern "C" void kernel_launch(void* const* d_in, const int* in_sizes, int n_in,
                              void* d_out, int out_size, void* d_ws, size_t ws_size,
                              hipStream_t stream)
{
    const float* x     = (const float*)d_in[0];
    const float* qkv_w = (const float*)d_in[1];
    const float* qkv_b = (const float*)d_in[2];
    const float* out_w = (const float*)d_in[3];
    const float* out_b = (const float*)d_in[4];
    const float* pre_g = (const float*)d_in[5];
    const float* pre_b = (const float*)d_in[6];
    const float* lcm_g = (const float*)d_in[7];
    const float* lcm_b = (const float*)d_in[8];
    const float* ci_w  = (const float*)d_in[9];
    const float* ci_b  = (const float*)d_in[10];
    const float* bn_g  = (const float*)d_in[11];
    const float* bn_b  = (const float*)d_in[12];
    const float* co_w  = (const float*)d_in[13];
    const float* co_b  = (const float*)d_in[14];
    float* out = (float*)d_out;
    char* ws = (char*)d_ws;

    _Float16* normed_h  = (_Float16*)(ws + OFF_NORMED);
    _Float16* ascaled_h = (_Float16*)(ws + OFF_NORMED);   // reuse after GEMM1
    _Float16* qkvw_h    = (_Float16*)(ws + OFF_QKVW_H);
    _Float16* outw_h    = (_Float16*)(ws + OFF_OUTW_H);
    float*    kv_f      = (float*)   (ws + OFF_KV_F);
    _Float16* kv_h      = (_Float16*)(ws + OFF_KV_H);
    float*    normzr    = (float*)   (ws + OFF_NORMZR);
    float*    partials  = (float*)   (ws + OFF_PART);
    float*    bnss      = (float*)   (ws + OFF_BNSS);
    _Float16* qkv_h     = (_Float16*)(ws + OFF_QKV_H);
    float*    attn1     = (float*)   (ws + OFF_ATTN1);
    float*    proj      = (float*)   (ws + OFF_PROJ);
    float*    xi        = (float*)   (ws + OFF_XI);
    float*    yb        = (float*)   (ws + OFF_Y);
    float*    y2        = (float*)   (ws + OFF_Y2);

    const long long sAb = (long long)Nt * QKVLD;   // per-image stride in qkv

    // 1) weights -> f16
    cvt_f32_f16_kernel<<<(786432 + 255)/256, 256, 0, stream>>>(qkv_w, qkvw_h, 786432);
    cvt_f32_f16_kernel<<<(262144 + 255)/256, 256, 0, stream>>>(out_w, outw_h, 262144);

    // 2) pre-LN -> f16
    layernorm_kernel<<<Bx*Nt, 256, 0, stream>>>(x, pre_g, pre_b, normed_h, nullptr, 0);

    // 3) QKV GEMM: [32768x512] x [512x1536] -> qkv f16
    gemm_wmma_kernel<128,128,64,2,4><<<dim3(12, 256, 1), 256, 0, stream>>>(
        normed_h, qkvw_h, qkv_b, (void*)qkv_h, 1,
        Bx*Nt, 1536, Cc, Cc, 1536, 1536,
        0, 0, 0, 0, 0, 0, 0, 0, 0, 1);

    // 4) normalizer = max(sum_n relu(k), 100)
    ksum_kernel<<<64, 256, 0, stream>>>(qkv_h, normzr);

    // 5) kv = relu(k)^T v : batched 64, M=N=64, K=4096 (transA)
    gemm_wmma_kernel<64,64,64,2,2><<<dim3(1, 1, 64), 128, 0, stream>>>(
        qkv_h + 512, qkv_h + 1024, nullptr, (void*)kv_f, 0,
        64, 64, Nt, (int)QKVLD, (int)QKVLD, 64,
        1, 1, 0,
        sAb, 64, sAb, 64, 8ll*4096, 4096, 8);
    cvt_f32_f16_kernel<<<(262144 + 255)/256, 256, 0, stream>>>(kv_f, kv_h, 262144);

    // 6) attn1 = relu(q) @ kv : batched 64, M=4096, N=64, K=64
    gemm_wmma_kernel<128,64,64,4,2><<<dim3(1, 32, 64), 256, 0, stream>>>(
        qkv_h, kv_h, nullptr, (void*)attn1, 0,
        Nt, 64, 64, (int)QKVLD, 64, 64,
        0, 1, 0,
        sAb, 64, 8ll*4096, 4096, 8ll*Nt*64, (long long)Nt*64, 8);

    // 7) scale by denom, reshape to [B,N,C] f16
    attn_scale_kernel<<<Bx*Nt, 512, 0, stream>>>(qkv_h, attn1, normzr, ascaled_h);

    // 8) output projection: [32768x512] x [512x512] -> proj f32 (+out_b)
    gemm_wmma_kernel<128,128,64,2,4><<<dim3(4, 256, 1), 256, 0, stream>>>(
        ascaled_h, outw_h, out_b, (void*)proj, 0,
        Bx*Nt, Cc, Cc, Cc, Cc, Cc,
        0, 0, 0, 0, 0, 0, 0, 0, 0, 1);

    // 9) LCM layer-norm, transpose to [B,C,H,W]
    layernorm_kernel<<<Bx*Nt, 256, 0, stream>>>(proj, lcm_g, lcm_b, nullptr, xi, 1);

    // 10) depthwise conv1 + GELU + partial BN stats
    dwconv_gelu_kernel<<<dim3(16, Cc, Bx), 256, 0, stream>>>(xi, ci_w, ci_b, yb, partials);

    // 11) BN finalize + apply
    bn_finalize_kernel<<<1, 512, 0, stream>>>(partials, bn_g, bn_b, bnss);
    bn_apply_kernel<<<65536, 256, 0, stream>>>(yb, bnss, y2);

    // 12) depthwise conv2 + residual -> out [B,N,C]
    dwconv_residual_kernel<<<dim3(16, Cc, Bx), 256, 0, stream>>>(y2, co_w, co_b, x, proj, out);
}
